// MultiHeadAttention_17798344474985
// MI455X (gfx1250) — matrix operand order repaired and checked by reading, not compiled, not run on hardware
//
#include <hip/hip_runtime.h>
#include <hip/hip_bf16.h>

// ---------------------------------------------------------------------------
// MHA forward for MI455X (gfx1250): bf16 WMMA (v_wmma_f32_16x16x32_bf16,
// f32 accumulate) for QKV GEMM, causal flash-attention, output projection.
// wave32 A/B operands use contiguous k = 16*half + j; C/D per ISA. GEMM inner loop is explicitly
// double-buffered across K-slabs so WMMA bursts overlap the next slab's loads.
// ---------------------------------------------------------------------------

#define BATCH 2
#define TSEQ  2048
#define CEMB  1024
#define NHEAD 16
#define HDIM  64

typedef __attribute__((ext_vector_type(16))) __bf16 v16bf;
typedef __attribute__((ext_vector_type(8)))  __bf16 bf16x8;
typedef __attribute__((ext_vector_type(8)))  float  v8f;

static __device__ inline v16bf make_v16(bf16x8 lo, bf16x8 hi) {
  v16bf r;
#pragma unroll
  for (int j = 0; j < 8; ++j) { r[j] = lo[j]; r[j + 8] = hi[j]; }
  return r;
}

static __device__ inline v8f wmma_bf16(v16bf a, v16bf b, v8f c) {
  return __builtin_amdgcn_wmma_f32_16x16x32_bf16(
      false, a, false, b, (short)0, c, false, false);
}

// ---------------------------------------------------------------------------
// f32 -> bf16 conversion
// ---------------------------------------------------------------------------
__global__ __launch_bounds__(256) void cvt_f32_bf16(
    const float* __restrict__ in, __bf16* __restrict__ out, int n) {
  int i = blockIdx.x * 256 + threadIdx.x;
  if (i < n) out[i] = (__bf16)in[i];
}

// ---------------------------------------------------------------------------
// out[m,n] = sum_k A[m,k] * W[n,k] + bias[n]
// A: [M,K] bf16 row-major. W: [N,K] bf16 row-major (torch Linear layout).
// One wave32 per 32(M) x 64(N) tile. Double-buffered K-slabs: while the
// 8-WMMA burst consumes slab k, the 12 b128 loads for slab k+1 are in
// flight (buffer parity is compile-time via unroll 2).
// grid = (N/256, M/32), block = 128 (4 waves -> 32M x 256N per block).
// ---------------------------------------------------------------------------
__global__ __launch_bounds__(128) void gemm_bf16_nt(
    const __bf16* __restrict__ A, const __bf16* __restrict__ W,
    const float* __restrict__ bias, float* __restrict__ out,
    int M, int N, int K) {
  const int lane = threadIdx.x & 31;
  const int wave = threadIdx.x >> 5;
  const int half = lane >> 4;   // wave32 half selects K sub-slab
  const int l16  = lane & 15;
  const int nbase = blockIdx.x * 256 + wave * 64;
  const int mbase = blockIdx.y * 32;

  const __bf16* arow0 = A + (size_t)(mbase + l16) * K;
  const __bf16* arow1 = A + (size_t)(mbase + 16 + l16) * K;
  const __bf16* wbase = W + (size_t)(nbase + l16) * K;

  v8f acc[2][4] = {};
  v16bf abuf[2][2];   // [parity][m-subtile]
  v16bf bbuf[2][4];   // [parity][n-subtile]

  auto load_slab = [&](int k0, int p) {
    // A operands (16x32): lane holds one row; k = j + 16*half
    abuf[p][0] = make_v16(*(const bf16x8*)(arow0 + k0 + 16 * half),
                          *(const bf16x8*)(arow0 + k0 + 16 * half + 8));
    abuf[p][1] = make_v16(*(const bf16x8*)(arow1 + k0 + 16 * half),
                          *(const bf16x8*)(arow1 + k0 + 16 * half + 8));
    // B operands (32x16): lane holds col; k = j + 16*half
#pragma unroll
    for (int nt = 0; nt < 4; ++nt) {
      const __bf16* wrow = wbase + (size_t)nt * 16 * K + k0 + 16 * half;
      bbuf[p][nt] = make_v16(*(const bf16x8*)(wrow), *(const bf16x8*)(wrow + 8));
    }
  };

  load_slab(0, 0);
#pragma unroll 2
  for (int k0 = 0; k0 < K; k0 += 32) {
    const int cur = (k0 >> 5) & 1;   // compile-time after unroll
    const int nxt = cur ^ 1;
    if (k0 + 32 < K) {
      load_slab(k0 + 32, nxt);       // in flight during this slab's WMMAs
      __builtin_prefetch(arow0 + k0 + 64, 0, 3);  // global_prefetch_b8, k+2
      __builtin_prefetch(arow1 + k0 + 64, 0, 3);
    }
#pragma unroll
    for (int nt = 0; nt < 4; ++nt) {
      acc[0][nt] = wmma_bf16(abuf[cur][0], bbuf[cur][nt], acc[0][nt]);
      acc[1][nt] = wmma_bf16(abuf[cur][1], bbuf[cur][nt], acc[1][nt]);
    }
  }
  // C/D layout: VGPR r -> row r + 8*half, lane%16 -> col
#pragma unroll
  for (int mt = 0; mt < 2; ++mt)
#pragma unroll
    for (int nt = 0; nt < 4; ++nt) {
      int n = nbase + nt * 16 + l16;
      float bz = bias[n];
#pragma unroll
      for (int r = 0; r < 8; ++r)
        out[(size_t)(mbase + mt * 16 + r + 8 * half) * N + n] =
            acc[mt][nt][r] + bz;
    }
}

// ---------------------------------------------------------------------------
// RoPE on q,k: qkv f32 [B,T,3C] (bias already added by GEMM) ->
//   Qr, Kr bf16 [B,H,T,D] (rotated). V handled by v_pack (transpose).
// One thread per (b,t,h,i), i in [0,32): the (d, d+32) rotation pair.
// ---------------------------------------------------------------------------
__global__ __launch_bounds__(256) void rope_pack(
    const float* __restrict__ qkv, __bf16* __restrict__ Qr,
    __bf16* __restrict__ Kr) {
  int idx = blockIdx.x * 256 + threadIdx.x;   // 2^21 threads
  int i = idx & 31;
  int h = (idx >> 5) & (NHEAD - 1);
  int t = (idx >> 9) & (TSEQ - 1);
  int b = idx >> 20;

  size_t row = ((size_t)b * TSEQ + t) * (3 * CEMB);
  int col = h * HDIM + i;
  float q0 = qkv[row + col],        q1 = qkv[row + col + 32];
  float k0 = qkv[row + CEMB + col], k1 = qkv[row + CEMB + col + 32];

  // inv_freq = 10000^(-i/32); ln(10000) = 9.210340371976184
  float inv = __expf(-(float)i * 0.03125f * 9.210340371976184f);
  float ang = (float)t * inv;
  float c = cosf(ang), s = sinf(ang);

  int bh = b * NHEAD + h;
  size_t qko = ((size_t)bh * TSEQ + t) * HDIM;
  Qr[qko + i]      = (__bf16)(q0 * c - q1 * s);
  Qr[qko + i + 32] = (__bf16)(q1 * c + q0 * s);
  Kr[qko + i]      = (__bf16)(k0 * c - k1 * s);
  Kr[qko + i + 32] = (__bf16)(k1 * c + k0 * s);
}

// ---------------------------------------------------------------------------
// V transpose via LDS tile: qkv v-part f32 [B,T,3C] -> Vt bf16 [B,H,D,T].
// Block handles a 64(t) x 32(d) tile; reads coalesced along d, writes
// 4-byte packed pairs coalesced along t. Padded LDS row kills conflicts.
// grid = (T/64, D/32, B*H), block = 256.
// ---------------------------------------------------------------------------
__global__ __launch_bounds__(256) void v_pack(
    const float* __restrict__ qkv, __bf16* __restrict__ Vt) {
  __shared__ __bf16 tile[64][34];
  const int lane = threadIdx.x & 31;
  const int grp  = threadIdx.x >> 5;           // 0..7
  const int tt0  = blockIdx.x * 64;
  const int dd0  = blockIdx.y * 32;
  const int bh   = blockIdx.z;
  const int b    = bh >> 4;
  const int h    = bh & (NHEAD - 1);

  // read: 8 rows of 32 d per thread-group, coalesced along d
#pragma unroll
  for (int tt = 0; tt < 8; ++tt) {
    int tl = grp * 8 + tt;
    size_t src = ((size_t)b * TSEQ + tt0 + tl) * (3 * CEMB) +
                 2 * CEMB + h * HDIM + dd0 + lane;
    tile[tl][lane] = (__bf16)qkv[src];
  }
  __syncthreads();

  // write: each thread stores packed bf16 pairs along t (4B, coalesced)
#pragma unroll
  for (int dd = 0; dd < 4; ++dd) {
    int dl = grp * 4 + dd;
    union { __bf16 hx[2]; unsigned int u; } pk;
    pk.hx[0] = tile[2 * lane][dl];
    pk.hx[1] = tile[2 * lane + 1][dl];
    size_t dst = ((size_t)bh * HDIM + dd0 + dl) * TSEQ + tt0 + 2 * lane;
    *(unsigned int*)(Vt + dst) = pk.u;
  }
}

// ---------------------------------------------------------------------------
// Flash attention, causal. One wave32 per (b,h, 32-query tile).
// Per 32-key block: 8 WMMAs for S (K-operands shared by both q-subtiles),
// online softmax (half-wave shfl_xor reductions), P re-layout via 2KB LDS
// tile, 8 WMMAs for P*V (V-operands shared).
// grid = (T/32, B*H), block = 32.
// ---------------------------------------------------------------------------
__global__ __launch_bounds__(32) void attn_fa(
    const __bf16* __restrict__ Q, const __bf16* __restrict__ K,
    const __bf16* __restrict__ Vt, __bf16* __restrict__ Yb) {
  __shared__ __bf16 lds_p[32][32];
  const int lane = threadIdx.x;
  const int half = lane >> 4;
  const int l16  = lane & 15;
  const int qt   = blockIdx.x;
  const int bh   = blockIdx.y;
  const int b    = bh >> 4;
  const int h    = bh & (NHEAD - 1);
  const int qbase = qt * 32;

  // Q A-operands: [q-subtile][d-chunk]; lane holds one query row
  v16bf qa[2][2];
#pragma unroll
  for (int qs = 0; qs < 2; ++qs) {
    const __bf16* qrow = Q + ((size_t)bh * TSEQ + qbase + qs * 16 + l16) * HDIM;
#pragma unroll
    for (int dc = 0; dc < 2; ++dc)
      qa[qs][dc] = make_v16(*(const bf16x8*)(qrow + dc * 32 + 16 * half),
                            *(const bf16x8*)(qrow + dc * 32 + 16 * half + 8));
  }

  float mrow[2][8], lrow[2][8];
  v8f acc[2][4] = {};
#pragma unroll
  for (int qs = 0; qs < 2; ++qs)
#pragma unroll
    for (int r = 0; r < 8; ++r) { mrow[qs][r] = -3.0e38f; lrow[qs][r] = 0.f; }

  for (int kb = 0; kb <= qbase; kb += 32) {
    // ---- S = Q K^T: 32 queries x 32 keys as 4 C-tiles (8 WMMAs) ----
    v8f s[2][2];                                   // [q-subtile][key-subtile]
#pragma unroll
    for (int st = 0; st < 2; ++st) {
      v8f c0 = {}, c1 = {};
#pragma unroll
      for (int dc = 0; dc < 2; ++dc) {
        const __bf16* krow = K + ((size_t)bh * TSEQ + kb + st * 16 + l16) * HDIM +
                             dc * 32 + 16 * half;
        v16bf kbop = make_v16(*(const bf16x8*)(krow), *(const bf16x8*)(krow + 8));
        c0 = wmma_bf16(qa[0][dc], kbop, c0);
        c1 = wmma_bf16(qa[1][dc], kbop, c1);
      }
      s[0][st] = c0; s[1][st] = c1;
    }
    // ---- scale, causal mask, online softmax (per q-subtile) ----
    float scl[2][8];
#pragma unroll
    for (int qs = 0; qs < 2; ++qs)
#pragma unroll
      for (int r = 0; r < 8; ++r) {
        int qi = qbase + qs * 16 + r + 8 * half;   // this lane's query row
        float v0 = s[qs][0][r] * 0.125f;           // 1/sqrt(64)
        float v1 = s[qs][1][r] * 0.125f;
        if (kb + l16 > qi)      v0 = -3.0e38f;
        if (kb + 16 + l16 > qi) v1 = -3.0e38f;
        float rm = fmaxf(v0, v1);
        rm = fmaxf(rm, __shfl_xor(rm, 1, 32));
        rm = fmaxf(rm, __shfl_xor(rm, 2, 32));
        rm = fmaxf(rm, __shfl_xor(rm, 4, 32));
        rm = fmaxf(rm, __shfl_xor(rm, 8, 32));     // reduce in 16-lane half
        float mnew = fmaxf(mrow[qs][r], rm);
        scl[qs][r] = __expf(mrow[qs][r] - mnew);
        float p0 = __expf(v0 - mnew);
        float p1 = __expf(v1 - mnew);
        s[qs][0][r] = p0; s[qs][1][r] = p1;
        float rs = p0 + p1;
        rs += __shfl_xor(rs, 1, 32);
        rs += __shfl_xor(rs, 2, 32);
        rs += __shfl_xor(rs, 4, 32);
        rs += __shfl_xor(rs, 8, 32);
        lrow[qs][r] = lrow[qs][r] * scl[qs][r] + rs;
        mrow[qs][r] = mnew;
      }
#pragma unroll
    for (int qs = 0; qs < 2; ++qs)
#pragma unroll
      for (int nt = 0; nt < 4; ++nt)
#pragma unroll
        for (int r = 0; r < 8; ++r) acc[qs][nt][r] *= scl[qs][r];

    // ---- re-layout P (C/D layout -> A layout) through LDS ----
    __syncthreads();
#pragma unroll
    for (int qs = 0; qs < 2; ++qs)
#pragma unroll
      for (int r = 0; r < 8; ++r) {
        lds_p[qs * 16 + r + 8 * half][l16]      = (__bf16)s[qs][0][r];
        lds_p[qs * 16 + r + 8 * half][16 + l16] = (__bf16)s[qs][1][r];
      }
    __syncthreads();
    v16bf pa[2];
#pragma unroll
    for (int qs = 0; qs < 2; ++qs)
      pa[qs] = make_v16(*(const bf16x8*)(&lds_p[qs * 16 + l16][16 * half]),
                        *(const bf16x8*)(&lds_p[qs * 16 + l16][16 * half + 8]));

    // ---- acc += P * V (V-operand shared by both q-subtiles) ----
#pragma unroll
    for (int nt = 0; nt < 4; ++nt) {
      const __bf16* vrow =
          Vt + ((size_t)bh * HDIM + nt * 16 + l16) * TSEQ + kb + 16 * half;
      v16bf vb = make_v16(*(const bf16x8*)(vrow), *(const bf16x8*)(vrow + 8));
      acc[0][nt] = wmma_bf16(pa[0], vb, acc[0][nt]);
      acc[1][nt] = wmma_bf16(pa[1], vb, acc[1][nt]);
    }
  }

  // normalize and store y as bf16 in [B,T,C] for the projection GEMM
#pragma unroll
  for (int qs = 0; qs < 2; ++qs)
#pragma unroll
    for (int nt = 0; nt < 4; ++nt)
#pragma unroll
      for (int r = 0; r < 8; ++r) {
        float y = acc[qs][nt][r] / lrow[qs][r];
        int t = qbase + qs * 16 + r + 8 * half;
        Yb[((size_t)b * TSEQ + t) * CEMB + h * HDIM + nt * 16 + l16] = (__bf16)y;
      }
}

// ---------------------------------------------------------------------------
extern "C" void kernel_launch(void* const* d_in, const int* in_sizes, int n_in,
                              void* d_out, int out_size, void* d_ws, size_t ws_size,
                              hipStream_t stream) {
  (void)in_sizes; (void)n_in; (void)out_size; (void)ws_size;
  const float* x      = (const float*)d_in[0];   // [B,T,C]
  const float* W_attn = (const float*)d_in[1];   // [3C,C]
  const float* b_attn = (const float*)d_in[2];   // [3C]
  const float* W_proj = (const float*)d_in[3];   // [C,C]
  const float* b_proj = (const float*)d_in[4];   // [C]
  float* out = (float*)d_out;                    // [B,T,C] f32

  const int M  = BATCH * TSEQ;                   // 4096
  const int C3 = 3 * CEMB;                       // 3072

  char* ws = (char*)d_ws;
  size_t off = 0;
  auto alloc = [&](size_t bytes) -> void* {
    void* p = ws + off;
    off += (bytes + 255) & ~(size_t)255;
    return p;
  };

  __bf16* xb  = (__bf16*)alloc((size_t)M * CEMB * 2);     // 8 MB
  __bf16* Wab = (__bf16*)alloc((size_t)C3 * CEMB * 2);    // 6 MB
  __bf16* Wpb = (__bf16*)alloc((size_t)CEMB * CEMB * 2);  // 2 MB
  float*  qkv = (float*)alloc((size_t)M * C3 * 4);        // 48 MB
  __bf16* Qr  = (__bf16*)alloc((size_t)M * CEMB * 2);     // 8 MB
  __bf16* Kr  = (__bf16*)alloc((size_t)M * CEMB * 2);     // 8 MB
  __bf16* Vt  = (__bf16*)alloc((size_t)M * CEMB * 2);     // 8 MB
  __bf16* Yb  = xb;  // xb dead after QKV GEMM; reuse for attention output

  // 1) bf16 copies of activations and weights
  cvt_f32_bf16<<<(M * CEMB) / 256, 256, 0, stream>>>(x, xb, M * CEMB);
  cvt_f32_bf16<<<(C3 * CEMB) / 256, 256, 0, stream>>>(W_attn, Wab, C3 * CEMB);
  cvt_f32_bf16<<<(CEMB * CEMB) / 256, 256, 0, stream>>>(W_proj, Wpb, CEMB * CEMB);

  // 2) qkv = x @ W_attn^T + b_attn   (f32 accumulate/out)
  gemm_bf16_nt<<<dim3(C3 / 256, M / 32), 128, 0, stream>>>(
      xb, Wab, b_attn, qkv, M, C3, CEMB);

  // 3) RoPE (q,k) and V transpose
  rope_pack<<<(BATCH * TSEQ * NHEAD * 32) / 256, 256, 0, stream>>>(qkv, Qr, Kr);
  v_pack<<<dim3(TSEQ / 64, HDIM / 32, BATCH * NHEAD), 256, 0, stream>>>(qkv, Vt);

  // 4) causal flash attention (32 queries per wave)
  attn_fa<<<dim3(TSEQ / 32, BATCH * NHEAD), 32, 0, stream>>>(Qr, Kr, Vt, Yb);

  // 5) out = y @ W_proj^T + b_proj
  gemm_bf16_nt<<<dim3(CEMB / 256, M / 32), 128, 0, stream>>>(
      Yb, Wpb, b_proj, out, M, CEMB, CEMB);
}
